// Attention_91010357002626
// MI455X (gfx1250) — matrix operand order repaired and checked by reading, not compiled, not run on hardware
//
#include <hip/hip_runtime.h>

// ---------------------------------------------------------------------------
// BiDAF attention for MI455X (gfx1250, wave32, WMMA).
// bf16 WMMA (16x16x32) with fp32 accumulate keeps the kernel HBM-bound
// (~900MB traffic @ 23.3TB/s ~= 39us); fp32 WMMA (K=4) would be compute-bound.
// ---------------------------------------------------------------------------

typedef __bf16 bf16;
typedef __attribute__((ext_vector_type(16))) __bf16 v16bf;
typedef __attribute__((ext_vector_type(8)))  __bf16 v8bf;
typedef __attribute__((ext_vector_type(8)))  float  v8f;

#define Nn      64
#define JXn     2048
#define JQn     128
#define Dd      256
#define TX      128      // JX rows per block
#define HS      264      // padded bf16 row stride (528B -> LDS bank step 4, conflict-free b128)
#define AS      136      // padded bf16 row stride for a / u^T tiles (272B -> bank step 4)
#define OUTW    (4*Dd)   // 1024 output cols per row
#define NEG_BIG 100000.0f
#define PART_STRIDE 272  // floats per block partial: [0]=L, [1]=S, [8..264)=V

struct __align__(16) Smem {
  bf16 u[JQn * HS];                         // 67584 B : u[q][d], bf16
  union {
    bf16 h[TX * HS];                        // 67584 B : h tile (phase 1 + V stage)
    struct {
      bf16 a[TX * AS];                      // 34816 B : softmax result (A-frag source)
      bf16 ut[128 * AS];                    // 34816 B : u^T half-tile (B-frag source)
    } p2;
  } un;
  float rowm[TX];
  float wgt[TX];
  float red[16];
};
static_assert(sizeof(Smem) < 160 * 1024, "keep LDS under possible CU-mode cap");

__device__ __forceinline__ v8f wmma_bf16(v16bf a, v16bf b, v8f c) {
  return __builtin_amdgcn_wmma_f32_16x16x32_bf16(false, a, false, b, (short)0, c,
                                                 false, false);
}

// A-frag: per lane, K chunks at +0 and +16 elements (ISA 16-bit 16x32 A layout)
__device__ __forceinline__ v16bf ldfragA(const bf16* p) {
  v8bf lo = *(const v8bf*)p;
  v8bf hi = *(const v8bf*)(p + 16);
  return __builtin_shufflevector(lo, hi, 0,1,2,3,4,5,6,7,8,9,10,11,12,13,14,15);
}
// B-frag: per lane, K chunks at +0 and +16 elements (same K order as A-frag)
__device__ __forceinline__ v16bf ldfragB(const bf16* p) {
  v8bf lo = *(const v8bf*)p;
  v8bf hi = *(const v8bf*)(p + 16);
  return __builtin_shufflevector(lo, hi, 0,1,2,3,4,5,6,7,8,9,10,11,12,13,14,15);
}

__global__ __launch_bounds__(256, 1)
void attn_main(const float* __restrict__ hg, const float* __restrict__ ug,
               const unsigned char* __restrict__ hmask,
               const unsigned char* __restrict__ umask,
               float* __restrict__ out, float* __restrict__ part) {
  __shared__ Smem sm;
  const int tid  = threadIdx.x;
  const int xt   = blockIdx.x;     // 0..15
  const int n    = blockIdx.y;     // 0..63
  const int x0   = xt * TX;
  const int wid  = tid >> 5;
  const int lane = tid & 31;
  const int half = lane >> 4;
  const int l16  = lane & 15;

  const float* hrow0 = hg + ((size_t)n * JXn + x0) * Dd;
  const float* urow0 = ug + (size_t)n * JQn * Dd;

  // ---- stage u and h as bf16 into LDS (coalesced float4 global reads) ----
  #pragma unroll 4
  for (int i = tid; i < JQn * (Dd / 4); i += 256) {
    int q = i >> 6, c = (i & 63) << 2;
    float4 f = *(const float4*)(urow0 + q * Dd + c);
    bf16* d = &sm.u[q * HS + c];
    d[0] = (bf16)f.x; d[1] = (bf16)f.y; d[2] = (bf16)f.z; d[3] = (bf16)f.w;
  }
  #pragma unroll 4
  for (int i = tid; i < TX * (Dd / 4); i += 256) {
    int r = i >> 6, c = (i & 63) << 2;
    float4 f = *(const float4*)(hrow0 + r * Dd + c);
    bf16* d = &sm.un.h[r * HS + c];
    d[0] = (bf16)f.x; d[1] = (bf16)f.y; d[2] = (bf16)f.z; d[3] = (bf16)f.w;
  }
  __syncthreads();

  // ---- phase 1: s = h_tile (128x256) @ u^T (256x128); wave = 16 rows x 128 cols
  v8f zero = {0.f, 0.f, 0.f, 0.f, 0.f, 0.f, 0.f, 0.f};
  v8f acc[8];
  #pragma unroll
  for (int nt = 0; nt < 8; ++nt) acc[nt] = zero;

  #pragma unroll
  for (int ks = 0; ks < 8; ++ks) {
    v16bf af = ldfragA(&sm.un.h[(wid * 16 + l16) * HS + ks * 32 + half * 8]);
    #pragma unroll
    for (int nt = 0; nt < 8; ++nt) {
      v16bf bfB = ldfragB(&sm.u[(nt * 16 + l16) * HS + ks * 32 + half * 8]);
      acc[nt] = wmma_bf16(af, bfB, acc[nt]);
    }
  }

  // ---- mask term: (h_mask & u_mask - 1) * NEG_BIG ----
  const unsigned char* hmp = hmask + (size_t)n * JXn + x0 + wid * 16 + half * 8;
  const unsigned char* ump = umask + (size_t)n * JQn;
  unsigned int hmr[8];
  #pragma unroll
  for (int r = 0; r < 8; ++r) hmr[r] = hmp[r];
  #pragma unroll
  for (int nt = 0; nt < 8; ++nt) {
    unsigned int ub = ump[nt * 16 + l16];
    #pragma unroll
    for (int r = 0; r < 8; ++r)
      acc[nt][r] += (hmr[r] & ub) ? 0.f : -NEG_BIG;
  }

  // ---- row softmax over JQ=128 (8 in-lane values + 16-lane butterfly) ----
  float mrow[8];
  #pragma unroll
  for (int r = 0; r < 8; ++r) {
    float m = acc[0][r];
    #pragma unroll
    for (int nt = 1; nt < 8; ++nt) m = fmaxf(m, acc[nt][r]);
    mrow[r] = m;
  }
  #pragma unroll
  for (int off = 1; off < 16; off <<= 1) {
    #pragma unroll
    for (int r = 0; r < 8; ++r)
      mrow[r] = fmaxf(mrow[r], __shfl_xor(mrow[r], off, 32));
  }
  float ls[8];
  #pragma unroll
  for (int r = 0; r < 8; ++r) ls[r] = 0.f;
  #pragma unroll
  for (int nt = 0; nt < 8; ++nt) {
    #pragma unroll
    for (int r = 0; r < 8; ++r) {
      float p = __expf(acc[nt][r] - mrow[r]);
      acc[nt][r] = p;
      ls[r] += p;
    }
  }
  #pragma unroll
  for (int off = 1; off < 16; off <<= 1) {
    #pragma unroll
    for (int r = 0; r < 8; ++r) ls[r] += __shfl_xor(ls[r], off, 32);
  }
  #pragma unroll
  for (int r = 0; r < 8; ++r) {
    float inv = 1.f / ls[r];
    #pragma unroll
    for (int nt = 0; nt < 8; ++nt) acc[nt][r] *= inv;
  }

  // ---- q2c partials: L = max rowm, S = sum exp(m-L), V[d] = sum exp(m-L)*h ----
  if (l16 == 0) {
    #pragma unroll
    for (int r = 0; r < 8; ++r) sm.rowm[wid * 16 + half * 8 + r] = mrow[r];
  }
  __syncthreads();

  float lv = (tid < TX) ? sm.rowm[tid] : -3.0e38f;
  #pragma unroll
  for (int off = 1; off < 32; off <<= 1) lv = fmaxf(lv, __shfl_xor(lv, off, 32));
  if (lane == 0) sm.red[wid] = lv;
  __syncthreads();
  float Lb = sm.red[0];
  #pragma unroll
  for (int i = 1; i < 8; ++i) Lb = fmaxf(Lb, sm.red[i]);

  float wv = 0.f;
  if (tid < TX) { wv = __expf(sm.rowm[tid] - Lb); sm.wgt[tid] = wv; }
  float sv = wv;
  #pragma unroll
  for (int off = 1; off < 32; off <<= 1) sv += __shfl_xor(sv, off, 32);
  if (lane == 0) sm.red[8 + wid] = sv;
  __syncthreads();
  float Sb = sm.red[8];
  #pragma unroll
  for (int i = 1; i < 8; ++i) Sb += sm.red[8 + i];

  // V partial: thread owns column d = tid (conflict-free LDS column walk)
  float vacc = 0.f;
  #pragma unroll 4
  for (int x = 0; x < TX; ++x) vacc += sm.wgt[x] * (float)sm.un.h[x * HS + tid];
  float* pb = part + (size_t)(n * 16 + xt) * PART_STRIDE;
  pb[8 + tid] = vacc;
  if (tid == 0) { pb[0] = Lb; pb[1] = Sb; }
  __syncthreads();   // h region now dead -> reuse as a / u^T

  // ---- round-trip softmax result through LDS into A-frag layout ----
  #pragma unroll
  for (int nt = 0; nt < 8; ++nt) {
    #pragma unroll
    for (int r = 0; r < 8; ++r)
      sm.un.p2.a[(wid * 16 + half * 8 + r) * AS + nt * 16 + l16] = (bf16)acc[nt][r];
  }

  // ---- phase 2: u_tilde = a (128x128) @ u (128x256), 2 passes of 128 cols ----
  float* outrow = out + ((size_t)n * JXn + x0) * OUTW;
  for (int pass = 0; pass < 2; ++pass) {
    // build u^T half tile: ut[dd][q] = u[q][pass*128+dd]
    for (int i = tid; i < JQn * 128; i += 256) {
      int q = i >> 7, dd = i & 127;
      sm.un.p2.ut[dd * AS + q] = sm.u[q * HS + pass * 128 + dd];
    }
    __syncthreads();

    v8f acc2[8];
    #pragma unroll
    for (int nt = 0; nt < 8; ++nt) acc2[nt] = zero;
    #pragma unroll
    for (int ks = 0; ks < 4; ++ks) {
      v16bf af = ldfragA(&sm.un.p2.a[(wid * 16 + l16) * AS + ks * 32 + half * 8]);
      #pragma unroll
      for (int nt = 0; nt < 8; ++nt) {
        v16bf bfB = ldfragB(&sm.un.p2.ut[(nt * 16 + l16) * AS + ks * 32 + half * 8]);
        acc2[nt] = wmma_bf16(af, bfB, acc2[nt]);
      }
    }
    #pragma unroll
    for (int nt = 0; nt < 8; ++nt) {
      #pragma unroll
      for (int r = 0; r < 8; ++r) {
        int row = wid * 16 + half * 8 + r;
        int c = pass * 128 + nt * 16 + l16;
        outrow[(size_t)row * OUTW + Dd + c] = acc2[nt][r];  // out block 1: u_tilde
      }
    }
    __syncthreads();  // ut reads done before next pass overwrites
  }
}

// combine 16 per-tile partials -> h_tilde[n][d]
__global__ __launch_bounds__(256)
void q2c_combine(const float* __restrict__ part, float* __restrict__ htld) {
  int n = blockIdx.x, d = threadIdx.x;
  float G = -3.0e38f;
  #pragma unroll
  for (int i = 0; i < 16; ++i)
    G = fmaxf(G, part[(size_t)(n * 16 + i) * PART_STRIDE]);
  float S = 0.f, V = 0.f;
  #pragma unroll
  for (int i = 0; i < 16; ++i) {
    const float* p = part + (size_t)(n * 16 + i) * PART_STRIDE;
    float sc = __expf(p[0] - G);
    S += p[1] * sc;
    V += p[8 + d] * sc;
  }
  htld[n * Dd + d] = V / S;
}

// out blocks 0 (h), 2 (h*u_tilde), 3 (h*h_tilde); float4 streaming
__global__ __launch_bounds__(256)
void finalize(const float* __restrict__ hg, const float* __restrict__ htld,
              float* __restrict__ out) {
  size_t g = (size_t)blockIdx.x * blockDim.x + threadIdx.x;  // one float4 each
  int j = (int)(g & 63);
  size_t row = g >> 6;              // n*JX + x
  int n = (int)(row >> 11);
  const float4 h4 = *(const float4*)(hg + row * Dd + j * 4);
  const float4 u4 = *(const float4*)(out + row * OUTW + Dd + j * 4);
  const float4 t4 = *(const float4*)(htld + (size_t)n * Dd + j * 4);
  float4 b2 = {h4.x * u4.x, h4.y * u4.y, h4.z * u4.z, h4.w * u4.w};
  float4 b3 = {h4.x * t4.x, h4.y * t4.y, h4.z * t4.z, h4.w * t4.w};
  *(float4*)(out + row * OUTW + 0 * Dd + j * 4) = h4;
  *(float4*)(out + row * OUTW + 2 * Dd + j * 4) = b2;
  *(float4*)(out + row * OUTW + 3 * Dd + j * 4) = b3;
}

extern "C" void kernel_launch(void* const* d_in, const int* in_sizes, int n_in,
                              void* d_out, int out_size, void* d_ws, size_t ws_size,
                              hipStream_t stream) {
  const float* h = (const float*)d_in[0];
  const float* u = (const float*)d_in[1];
  const unsigned char* hm = (const unsigned char*)d_in[2];
  const unsigned char* um = (const unsigned char*)d_in[3];
  float* out = (float*)d_out;

  float* part = (float*)d_ws;                       // 1024 * 272 floats
  float* htld = part + (size_t)Nn * 16 * PART_STRIDE; // 64 * 256 floats

  dim3 g1(JXn / TX, Nn);                    // 16 x 64 blocks
  attn_main<<<g1, 256, 0, stream>>>(h, u, hm, um, out, part);
  q2c_combine<<<Nn, 256, 0, stream>>>(part, htld);
  size_t nvec4 = (size_t)Nn * JXn * (Dd / 4);        // 8,388,608
  finalize<<<(unsigned)(nvec4 / 256), 256, 0, stream>>>(h, htld, out);
}
